// MultiHeadAttention_89369679495670
// MI455X (gfx1250) — matrix operand order repaired and checked by reading, not compiled, not run on hardware
//
#include <hip/hip_runtime.h>
#include <hip/hip_bf16.h>

// ---------------------------------------------------------------------------
// MHA for B=2, S=4096, D=512, H=8 (depth=64) on gfx1250 (wave32, WMMA).
// Pipeline: 3x GEMM (f16 WMMA, f32 acc) -> flash attention (WMMA) -> GEMM.
// ---------------------------------------------------------------------------

typedef __attribute__((ext_vector_type(16))) _Float16 v16h;
typedef __attribute__((ext_vector_type(8)))  _Float16 v8h;
typedef __attribute__((ext_vector_type(4)))  _Float16 v4h;
typedef __attribute__((ext_vector_type(2)))  _Float16 v2h;
typedef __attribute__((ext_vector_type(8)))  float    v8f;
typedef __attribute__((ext_vector_type(4)))  float    v4f;
typedef __attribute__((ext_vector_type(4)))  int      v4i;

union V16U { v16h v; v8h h[2]; };

constexpr int Bb    = 2;
constexpr int Sq    = 4096;
constexpr int Dm    = 512;
constexpr int Hh    = 8;
constexpr int DEPTH = 64;
constexpr int Mrows = Bb * Sq;          // 8192

constexpr int LDA = 40;  // 80 B rows  (16B aligned)
constexpr int LDW = 40;
constexpr int LDK = 72;  // 144 B rows (16B aligned)

#define AS1 __attribute__((address_space(1)))
#define AS3 __attribute__((address_space(3)))

#if defined(__AMDGCN__) && __has_builtin(__builtin_amdgcn_global_load_async_to_lds_b128)
#define HAVE_ASYNC_COPY 1
#else
#define HAVE_ASYNC_COPY 0
#endif

#if HAVE_ASYNC_COPY
// One 16-byte lane-wise async copy: LDS[dst] = MEM[src], tracked by ASYNCcnt.
__device__ inline void async_cp16(const _Float16* src, _Float16* lds_dst) {
  __builtin_amdgcn_global_load_async_to_lds_b128((AS1 v4i*)src,
                                                 (AS3 v4i*)lds_dst, 0, 0);
}
__device__ inline void async_wait0() {
#if __has_builtin(__builtin_amdgcn_s_wait_asynccnt)
  __builtin_amdgcn_s_wait_asynccnt(0);
#else
  asm volatile("s_wait_asynccnt 0x0" ::: "memory");
#endif
}
#endif

// ---- 16-lane row reductions via ds_swizzle (xor within lane groups) -------
template <int PAT>
__device__ inline float swzf(float x) {
  int i = __float_as_int(x);
  i = __builtin_amdgcn_ds_swizzle(i, PAT);
  return __int_as_float(i);
}
__device__ inline float rowmax16(float v) {
  v = fmaxf(v, swzf<0x7c01>(v));
  v = fmaxf(v, swzf<0x7c02>(v));
  v = fmaxf(v, swzf<0x7c04>(v));
  v = fmaxf(v, swzf<0x7c08>(v));
  return v;
}
__device__ inline float rowsum16(float v) {
  v += swzf<0x7c01>(v);
  v += swzf<0x7c02>(v);
  v += swzf<0x7c04>(v);
  v += swzf<0x7c08>(v);
  return v;
}

__device__ inline v8f wmma_f16(const V16U& a, const V16U& b, v8f c) {
  return __builtin_amdgcn_wmma_f32_16x16x32_f16(false, a.v, false, b.v,
                                                (short)0, c, false, false);
}

// load one 16x16 half-tile fragment chunk (two 16B LDS reads)
__device__ inline V16U ld_frag(const _Float16* p, int stride2) {
  V16U f;
  f.h[0] = *(const v8h*)(p);
  f.h[1] = *(const v8h*)(p + stride2);
  return f;
}

// ---------------------------------------------------------------------------
// GEMM: out[64x64 tile] = A[M x 512] * W[512 x 512] + bias
//   A_F16  : A is _Float16 (attention output) vs float (x1/x2)
//   HEADED : store f16 head-split [B,H,S,64] (QKV) vs flat f32 [M,512]
// Block: 128 threads (4 waves); wave w owns rows 16w..16w+15 of the tile.
// ---------------------------------------------------------------------------
template <bool A_F16, bool HEADED>
__global__ __launch_bounds__(128)
void gemm_bias_kernel(const void* __restrict__ Aptr,
                      const float* __restrict__ W,
                      const float* __restrict__ bias,
                      void* __restrict__ outPtr) {
  __shared__ _Float16 aLds[64 * LDA];
  __shared__ _Float16 wLds[64 * LDW];

  const int tid    = threadIdx.x;
  const int wave   = tid >> 5;
  const int lane   = tid & 31;
  const int ln16   = lane & 15;
  const int hiHalf = lane >> 4;
  const int mBase  = blockIdx.y * 64;
  const int nBase  = blockIdx.x * 64;

  v8f acc[4];
#pragma unroll
  for (int g = 0; g < 4; ++g) { v8f z = {}; acc[g] = z; }

  for (int k0 = 0; k0 < Dm; k0 += 32) {
    // ---- stage A tile (64 rows x 32 k) as f16, row-major ----
    if constexpr (A_F16) {
      const _Float16* A16 = (const _Float16*)Aptr;
#pragma unroll
      for (int i = 0; i < 2; ++i) {
        int slot = tid + i * 128;              // 256 slots x 8 halves
        int row  = slot >> 2;
        int off  = (slot & 3) * 8;
        const _Float16* src = A16 + (size_t)(mBase + row) * Dm + k0 + off;
        _Float16*       dst = aLds + row * LDA + off;
#if HAVE_ASYNC_COPY
        async_cp16(src, dst);                  // ASYNCcnt path, no VGPR hop
#else
        *(v8h*)dst = *(const v8h*)src;
#endif
      }
    } else {
      const float* A32 = (const float*)Aptr;
#pragma unroll
      for (int i = 0; i < 4; ++i) {
        int slot = tid + i * 128;              // 512 slots x 4 floats
        int row  = slot >> 3;
        int off  = (slot & 7) * 4;
        v4f d = *(const v4f*)(A32 + (size_t)(mBase + row) * Dm + k0 + off);
        v4h hv;
        hv.x = (_Float16)d.x; hv.y = (_Float16)d.y;
        hv.z = (_Float16)d.z; hv.w = (_Float16)d.w;
        *(v4h*)(aLds + row * LDA + off) = hv;
      }
    }
    // ---- stage W tile (32 k x 64 n), TRANSPOSED into wLds[n][k];
    //      two k-rows per slot so stores pack into ds_store_b32 ----
#pragma unroll
    for (int i = 0; i < 2; ++i) {
      int slot = tid + i * 128;                // 256 slots
      int krow = (slot >> 4) * 2;
      int noff = (slot & 15) * 4;
      v4f d0 = *(const v4f*)(W + (size_t)(k0 + krow) * Dm + nBase + noff);
      v4f d1 = *(const v4f*)(W + (size_t)(k0 + krow + 1) * Dm + nBase + noff);
#pragma unroll
      for (int j = 0; j < 4; ++j) {
        v2h t; t.x = (_Float16)d0[j]; t.y = (_Float16)d1[j];
        *(v2h*)(wLds + (noff + j) * LDW + krow) = t;
      }
    }
#if HAVE_ASYNC_COPY
    if constexpr (A_F16) async_wait0();
#endif
    __syncthreads();

    // ---- preload A fragment + all 4 B fragments (distinct registers so
    //      the ds_loads clause up and overlap with the WMMA chain) ----
    V16U af = ld_frag(aLds + (wave * 16 + ln16) * LDA + hiHalf * 8, 16);
    V16U bf[4];
#pragma unroll
    for (int g = 0; g < 4; ++g)
      bf[g] = ld_frag(wLds + (g * 16 + ln16) * LDW + hiHalf * 8, 16);
#pragma unroll
    for (int g = 0; g < 4; ++g) acc[g] = wmma_f16(af, bf[g], acc[g]);
    __syncthreads();
  }

  // ---- epilogue: bias + store ----
#pragma unroll
  for (int g = 0; g < 4; ++g) {
    int   ncol = nBase + g * 16 + ln16;
    float bv   = bias[ncol];
#pragma unroll
    for (int j = 0; j < 8; ++j) {
      int   mrow = mBase + wave * 16 + j + hiHalf * 8;
      float val  = acc[g][j] + bv;
      if constexpr (HEADED) {
        int b = mrow >> 12, s = mrow & (Sq - 1);
        int h = ncol >> 6,  dep = ncol & 63;
        _Float16* o16 = (_Float16*)outPtr;
        o16[(((size_t)b * Hh + h) * Sq + s) * DEPTH + dep] = (_Float16)val;
      } else {
        ((float*)outPtr)[(size_t)mrow * Dm + ncol] = val;
      }
    }
  }
}

// ---------------------------------------------------------------------------
// Flash attention: block = (64 q rows) x (one b,h); 4 waves x 16 rows each.
// Streams 64-key tiles: K -> LDS [key][depth] (async copy), V -> LDS
// transposed [depth][key]; online softmax with ds_swizzle row reductions;
// P goes through per-wave LDS to reshape into the WMMA A-fragment layout.
// ---------------------------------------------------------------------------
__global__ __launch_bounds__(128)
void flash_attn_kernel(const _Float16* __restrict__ Qh,
                       const _Float16* __restrict__ Kh,
                       const _Float16* __restrict__ Vh,
                       const int* __restrict__ mask,
                       _Float16* __restrict__ attnOut) {
  __shared__ _Float16 kLds[64 * LDK];
  __shared__ _Float16 vLds[64 * LDK];        // transposed: [depth][key]
  __shared__ _Float16 pLds[4][16 * LDK];     // per-wave P staging

  const int tid    = threadIdx.x;
  const int wave   = tid >> 5;
  const int lane   = tid & 31;
  const int ln16   = lane & 15;
  const int hiHalf = lane >> 4;

  const int    bh      = blockIdx.y;
  const int    b       = bh >> 3;
  const size_t headOff = (size_t)bh * Sq * DEPTH;
  const int    hIdx    = bh & 7;
  const int    qBase   = blockIdx.x * 64;

  // ---- load Q fragments once (2 depth chunks of 32) ----
  V16U qf[2];
  {
    const _Float16* qp =
        Qh + headOff + (size_t)(qBase + wave * 16 + ln16) * DEPTH;
#pragma unroll
    for (int c = 0; c < 2; ++c) {
      qf[c].h[0] = *(const v8h*)(qp + c * 32 + hiHalf * 8);
      qf[c].h[1] = *(const v8h*)(qp + c * 32 + hiHalf * 8 + 16);
    }
  }

  float m_i[8], l_i[8];
  v8f   accO[4];
#pragma unroll
  for (int j = 0; j < 8; ++j) { m_i[j] = -3.0e38f; l_i[j] = 0.0f; }
#pragma unroll
  for (int g = 0; g < 4; ++g) { v8f z = {}; accO[g] = z; }

  for (int kt0 = 0; kt0 < Sq; kt0 += 64) {
    // ---- stage K tile [key][depth] (async global->LDS if available) ----
#pragma unroll
    for (int i = 0; i < 4; ++i) {
      int slot = tid + i * 128;                // 512 slots x 8 halves
      int row  = slot >> 3;
      int off  = (slot & 7) * 8;
      const _Float16* src = Kh + headOff + (size_t)(kt0 + row) * DEPTH + off;
      _Float16*       dst = kLds + row * LDK + off;
#if HAVE_ASYNC_COPY
      async_cp16(src, dst);
#else
      *(v8h*)dst = *(const v8h*)src;
#endif
    }
    // ---- stage V tile transposed [depth][key]; two keys per slot so
    //      the transposed stores pack into ds_store_b32 ----
#pragma unroll
    for (int i = 0; i < 2; ++i) {
      int slot = tid + i * 128;                // 256 slots
      int key  = (slot >> 3) * 2;
      int off  = (slot & 7) * 8;
      const _Float16* vp = Vh + headOff + (size_t)(kt0 + key) * DEPTH + off;
      v8h d0 = *(const v8h*)(vp);
      v8h d1 = *(const v8h*)(vp + DEPTH);
#pragma unroll
      for (int j = 0; j < 8; ++j) {
        v2h t; t.x = d0[j]; t.y = d1[j];
        *(v2h*)(vLds + (off + j) * LDK + key) = t;
      }
    }
    if (kt0 + 64 < Sq)  // hint next K tile toward the caches
      __builtin_prefetch(Kh + headOff + (size_t)(kt0 + 64) * DEPTH + tid * 8,
                         0, 1);
#if HAVE_ASYNC_COPY
    async_wait0();
#endif
    __syncthreads();

    // ---- scores: preload all 8 K fragments, then 8 WMMAs ----
    V16U bk[4][2];
#pragma unroll
    for (int kt = 0; kt < 4; ++kt)
#pragma unroll
      for (int cch = 0; cch < 2; ++cch)
        bk[kt][cch] = ld_frag(
            kLds + (kt * 16 + ln16) * LDK + cch * 32 + hiHalf * 8, 16);

    v8f sc[4];
#pragma unroll
    for (int kt = 0; kt < 4; ++kt) {
      v8f c = {};
      c = wmma_f16(qf[0], bk[kt][0], c);
      c = wmma_f16(qf[1], bk[kt][1], c);
      int   kcol = kt0 + kt * 16 + ln16;
      float madd = (float)mask[b * Sq + kcol] * -1.0e9f;
#pragma unroll
      for (int j = 0; j < 8; ++j) c[j] = c[j] * 0.125f + madd;
      sc[kt] = c;
    }

    // ---- online softmax update (per row j; rows span 16 lanes) ----
#pragma unroll
    for (int j = 0; j < 8; ++j) {
      float tmax = fmaxf(fmaxf(sc[0][j], sc[1][j]), fmaxf(sc[2][j], sc[3][j]));
      tmax = rowmax16(tmax);
      float mnew = fmaxf(m_i[j], tmax);
      float sf   = __expf(m_i[j] - mnew);
      m_i[j]     = mnew;
      float rsum = 0.0f;
      int   prow = j + hiHalf * 8;
#pragma unroll
      for (int kt = 0; kt < 4; ++kt) {
        float e = __expf(sc[kt][j] - mnew);
        rsum += e;
        pLds[wave][prow * LDK + kt * 16 + ln16] = (_Float16)e;
      }
      l_i[j] = l_i[j] * sf + rowsum16(rsum);
#pragma unroll
      for (int g = 0; g < 4; ++g) accO[g][j] *= sf;
    }

    // ---- P (16x64) x V (64x64): per key-chunk preload P + 4 V frags ----
#pragma unroll
    for (int kc = 0; kc < 2; ++kc) {
      V16U pf =
          ld_frag(&pLds[wave][ln16 * LDK + kc * 32 + hiHalf * 8], 16);
      V16U vf[4];
#pragma unroll
      for (int g = 0; g < 4; ++g)
        vf[g] = ld_frag(
            vLds + (g * 16 + ln16) * LDK + kc * 32 + hiHalf * 8, 16);
#pragma unroll
      for (int g = 0; g < 4; ++g) accO[g] = wmma_f16(pf, vf[g], accO[g]);
    }
    __syncthreads();
  }

  // ---- normalize & store to [B, S, D] (heads re-interleaved) ----
#pragma unroll
  for (int j = 0; j < 8; ++j) {
    float r    = 1.0f / l_i[j];
    int   srow = qBase + wave * 16 + j + hiHalf * 8;
#pragma unroll
    for (int g = 0; g < 4; ++g) {
      int col = hIdx * DEPTH + g * 16 + ln16;
      attnOut[((size_t)b * Sq + srow) * Dm + col] = (_Float16)(accO[g][j] * r);
    }
  }
}

// ---------------------------------------------------------------------------
extern "C" void kernel_launch(void* const* d_in, const int* in_sizes, int n_in,
                              void* d_out, int out_size, void* d_ws,
                              size_t ws_size, hipStream_t stream) {
  const float* x1   = (const float*)d_in[0];
  const float* x2   = (const float*)d_in[1];
  const int*   mask = (const int*)d_in[2];
  const float* Wq   = (const float*)d_in[3];
  const float* bq   = (const float*)d_in[4];
  const float* Wk   = (const float*)d_in[5];
  const float* bk   = (const float*)d_in[6];
  const float* Wv   = (const float*)d_in[7];
  const float* bv   = (const float*)d_in[8];
  const float* Wo   = (const float*)d_in[9];
  const float* bo   = (const float*)d_in[10];

  const size_t elems = (size_t)Mrows * Dm;     // 4,194,304 f16 each
  _Float16* Qh = (_Float16*)d_ws;
  _Float16* Kh = Qh + elems;
  _Float16* Vh = Kh + elems;
  _Float16* Ah = Vh + elems;                   // attention output [B,S,D] f16

  dim3 blk(128);
  dim3 gridG(Dm / 64, Mrows / 64);             // (8, 128)
  gemm_bias_kernel<false, true><<<gridG, blk, 0, stream>>>(x1, Wq, bq, Qh);
  gemm_bias_kernel<false, true><<<gridG, blk, 0, stream>>>(x2, Wk, bk, Kh);
  gemm_bias_kernel<false, true><<<gridG, blk, 0, stream>>>(x2, Wv, bv, Vh);

  dim3 gridA(Sq / 64, Bb * Hh);                // (64, 16)
  flash_attn_kernel<<<gridA, blk, 0, stream>>>(Qh, Kh, Vh, mask, Ah);

  gemm_bias_kernel<true, false><<<gridG, blk, 0, stream>>>(Ah, Wo, bo,
                                                           (float*)d_out);
}
